// LlamaAttention_41936060678390
// MI455X (gfx1250) — matrix operand order repaired and checked by reading, not compiled, not run on hardware
//
#include <hip/hip_runtime.h>
#include <cstddef>

// ---------------------------------------------------------------------------
// MI455X (gfx1250) Llama attention, bf16 WMMA path + Tensor Data Mover.
//   - all 4 matmuls via v_wmma_f32_16x16x32_bf16 (wave32, 16x16x32 tiles)
//   - GEMM: A-tile (32 rows x 256 K) staged to LDS by TDM (tensor_load_to_lds),
//     double-buffered on TENSORcnt; shared by all 8 waves (8x A reuse),
//     2-way M register blocking (2x B reuse). Falls back to direct VMEM loads
//     if the TDM builtin is unavailable.
//   - attention: flash-style 2-pass softmax (probs written exactly once),
//     deterministic barrier-ordered cross-wave reduction.
// Workspace layout (bf16): hidden, Wq, Wk, Wv, Wo, q, k, v, attn  (~155 MB).
// ---------------------------------------------------------------------------

#define B_    2
#define S_    2048
#define H_    3584
#define NH_   28
#define NKV_  4
#define HD_   128
#define GROUPS_ 7
#define SCALING_ 0.0883883476483184f
#define NEG_  (-1.0e9f)

#define TILE_M 32      // GEMM rows per workgroup
#define KCH    256     // GEMM K-chunk staged per TDM transfer

typedef __attribute__((ext_vector_type(8)))  float    v8f;
typedef __attribute__((ext_vector_type(8)))  __bf16   v8bf;
typedef __attribute__((ext_vector_type(16))) __bf16   v16bf;
typedef __attribute__((ext_vector_type(4)))  unsigned u32x4;
typedef __attribute__((ext_vector_type(4)))  int      i32x4;
typedef __attribute__((ext_vector_type(8)))  int      i32x8;

#if defined(__gfx1250__) && __has_builtin(__builtin_amdgcn_tensor_load_to_lds)
#define HAVE_TDM 1
#else
#define HAVE_TDM 0
#endif
#if __has_include(<hip/amd_detail/amd_gfx1250_TDM.h>)
#define TDM_6ARG 1      // therock headers present -> clang-23 6-arg builtin
#else
#define TDM_6ARG 0      // ROCm 7.2 clang-22 5-arg builtin
#endif

__device__ __forceinline__ __bf16 f2bf(float x) {
  union { float f; unsigned u; } v; v.f = x;
  unsigned r = (v.u + 0x7fffu + ((v.u >> 16) & 1u)) >> 16;
  unsigned short hs = (unsigned short)r;
  __bf16 o; __builtin_memcpy(&o, &hs, 2);
  return o;
}
__device__ __forceinline__ float bf2f(__bf16 x) {
  unsigned short hs; __builtin_memcpy(&hs, &x, 2);
  union { unsigned u; float f; } v; v.u = ((unsigned)hs) << 16;
  return v.f;
}
__device__ __forceinline__ v8f vzero8() {
  v8f r;
  #pragma unroll
  for (int i = 0; i < 8; ++i) r[i] = 0.0f;
  return r;
}
__device__ __forceinline__ unsigned rfl(unsigned x) {
  return (unsigned)__builtin_amdgcn_readfirstlane((int)x);
}

// A fragment: 16x32 bf16, row-major source with leading dim ld (LDS or global).
__device__ __forceinline__ v16bf load_a16x32(const __bf16* base, int ld, int m0, int k0) {
  int lane = threadIdx.x & 31;
  const __bf16* p = base + (size_t)(m0 + (lane & 15)) * ld + k0 + ((lane >> 4) << 4);
  v8bf lo = *(const v8bf*)p;
  v8bf hi = *(const v8bf*)(p + 8);
  v16bf r;
  #pragma unroll
  for (int i = 0; i < 8; ++i) { r[i] = lo[i]; r[i + 8] = hi[i]; }
  return r;
}

// B fragment: 32x16 (KxN); source W[n,k] row-major (k contiguous).
__device__ __forceinline__ v16bf load_b32x16(const __bf16* base, int ld, int n0, int k0) {
  int lane = threadIdx.x & 31;
  const __bf16* p = base + (size_t)(n0 + (lane & 15)) * ld + k0 + ((lane >> 4) << 4);
  return *(const v16bf*)p;
}

#if HAVE_TDM
// TDM: stage a 2D bf16 tile (tile_m rows x tile_k cols, row stride ld elems)
// from global to LDS. Descriptor per CDNA5 ISA ch.8 (type=2, data_size=2B).
__device__ __forceinline__ void tdm_load_2d(unsigned lds_off, const __bf16* gptr,
                                            int ld, int tile_k, int tile_m) {
  unsigned long long ga = (unsigned long long)(size_t)gptr;
  u32x4 g0;
  g0[0] = rfl(1u);                                   // count=1, user mode
  g0[1] = rfl(lds_off);                              // lds_addr (bytes)
  g0[2] = rfl((unsigned)ga);                         // global_addr[31:0]
  g0[3] = rfl((unsigned)((ga >> 32) & 0x1FFFFFFu) | (2u << 30)); // addr hi | type=2
  unsigned dim0 = (unsigned)ld;                      // tensor_dim0 (elems)
  unsigned dim1 = 0x100000u;                         // tensor_dim1: large, no clip
  i32x8 g1;
  g1[0] = (int)rfl(1u << 16);                        // data_size=1 (2 bytes)
  g1[1] = (int)rfl((dim0 & 0xFFFFu) << 16);          // dim0 lo16 (bits 63:48)
  g1[2] = (int)rfl(((dim0 >> 16) & 0xFFFFu) | ((dim1 & 0xFFFFu) << 16));
  g1[3] = (int)rfl(((dim1 >> 16) & 0xFFFFu) | (((unsigned)tile_k & 0xFFFFu) << 16));
  g1[4] = (int)rfl((unsigned)tile_m & 0xFFFFu);      // tile_dim1; tile_dim2=0
  g1[5] = (int)rfl(dim0);                            // tensor_dim0_stride lo32
  g1[6] = 0;
  g1[7] = 0;
  i32x4 z4 = {0, 0, 0, 0};
#if TDM_6ARG
  i32x8 z8 = {0, 0, 0, 0, 0, 0, 0, 0};
  __builtin_amdgcn_tensor_load_to_lds(g0, g1, z4, z4, z8, 0);
#else
  __builtin_amdgcn_tensor_load_to_lds(g0, g1, z4, z4, 0);
#endif
}
#endif

// --------------------------- f32 -> bf16 convert ---------------------------
__global__ void cvt_kernel(const float* __restrict__ in, __bf16* __restrict__ out, size_t n) {
  size_t i = (size_t)blockIdx.x * blockDim.x + threadIdx.x;
  size_t stride = (size_t)gridDim.x * blockDim.x;
  for (; i < n; i += stride) out[i] = f2bf(in[i]);
}

// ------------------------------- GEMM ---------------------------------------
// C[M,N] = A[M,K] * W[N,K]^T (+ bias). 256 thr (8 waves); block tile 32x128;
// wave tile 32x16 (two WMMA accumulators). A staged to LDS by TDM, 2 buffers.
__global__ __launch_bounds__(256)
void gemm_kernel(const __bf16* __restrict__ A, const __bf16* __restrict__ W,
                 const float* __restrict__ bias,
                 __bf16* __restrict__ Cb, float* __restrict__ Cf,
                 int M, int N, int K) {
  __shared__ __bf16 lds_a[2][TILE_M * KCH];   // 2 x 16 KB

  const int wave = threadIdx.x >> 5;
  const int lane = threadIdx.x & 31;
  const int l15 = lane & 15, lh = lane >> 4;
  const int n0 = (blockIdx.x * 8 + wave) * 16;
  const int m0 = blockIdx.y * TILE_M;

  v8f acc0 = vzero8(), acc1 = vzero8();

#if HAVE_TDM
  const int nch = K / KCH;
  if (threadIdx.x < 32)
    tdm_load_2d((unsigned)(size_t)&lds_a[0][0], A + (size_t)m0 * K, K, KCH, TILE_M);

  for (int c = 0; c < nch; ++c) {
    if (threadIdx.x < 32) {
      if (c + 1 < nch) {
        tdm_load_2d((unsigned)(size_t)&lds_a[(c + 1) & 1][0],
                    A + (size_t)m0 * K + (size_t)(c + 1) * KCH, K, KCH, TILE_M);
        __builtin_amdgcn_s_wait_tensorcnt(1);   // chunk c landed
      } else {
        __builtin_amdgcn_s_wait_tensorcnt(0);
      }
    }
    __syncthreads();
    if (c + 1 < nch)   // pull next B chunk toward L2/L0 while we compute
      __builtin_prefetch(W + (size_t)(n0 + l15) * K + (size_t)(c + 1) * KCH, 0, 0);

    const __bf16* As = &lds_a[c & 1][0];
    #pragma unroll
    for (int kk = 0; kk < KCH; kk += 32) {
      v16bf a0 = load_a16x32(As, KCH, 0, kk);
      v16bf a1 = load_a16x32(As, KCH, 16, kk);
      v16bf b  = load_b32x16(W, K, n0, c * KCH + kk);
      acc0 = __builtin_amdgcn_wmma_f32_16x16x32_bf16(false, a0, false, b, (short)0, acc0, false, false);
      acc1 = __builtin_amdgcn_wmma_f32_16x16x32_bf16(false, a1, false, b, (short)0, acc1, false, false);
    }
    __syncthreads();
  }
#else
  #pragma unroll 4
  for (int k0 = 0; k0 < K; k0 += 32) {
    v16bf a0 = load_a16x32(A, K, m0, k0);
    v16bf a1 = load_a16x32(A, K, m0 + 16, k0);
    v16bf b  = load_b32x16(W, K, n0, k0);
    acc0 = __builtin_amdgcn_wmma_f32_16x16x32_bf16(false, a0, false, b, (short)0, acc0, false, false);
    acc1 = __builtin_amdgcn_wmma_f32_16x16x32_bf16(false, a1, false, b, (short)0, acc1, false, false);
  }
#endif

  int col = n0 + l15;
  float bv = bias ? bias[col] : 0.0f;
  #pragma unroll
  for (int i = 0; i < 8; ++i) {
    int row0 = m0 + lh * 8 + i;
    int row1 = row0 + 16;
    float v0 = acc0[i] + bv;
    float v1 = acc1[i] + bv;
    if (Cb) {
      Cb[(size_t)row0 * N + col] = f2bf(v0);
      Cb[(size_t)row1 * N + col] = f2bf(v1);
    } else {
      Cf[(size_t)row0 * N + col] = v0;
      Cf[(size_t)row1 * N + col] = v1;
    }
  }
}

// ------------------------------- RoPE ---------------------------------------
__global__ void rope_kernel(__bf16* __restrict__ x, const float* __restrict__ cosp,
                            const float* __restrict__ sinp, int nheads, long total) {
  long t = (long)blockIdx.x * blockDim.x + threadIdx.x;
  if (t >= total) return;
  int d = (int)(t & 63);
  long r = t >> 6;
  int hh = (int)(r % nheads); r /= nheads;
  int s  = (int)(r % S_);
  int bb = (int)(r / S_);
  size_t base = (((size_t)bb * S_ + s) * nheads + hh) * HD_;
  size_t cb   = ((size_t)bb * S_ + s) * HD_;
  float x1 = bf2f(x[base + d]);
  float x2 = bf2f(x[base + d + 64]);
  float c1 = cosp[cb + d],      s1 = sinp[cb + d];
  float c2 = cosp[cb + d + 64], s2 = sinp[cb + d + 64];
  x[base + d]      = f2bf(x1 * c1 - x2 * s1);
  x[base + d + 64] = f2bf(x2 * c2 + x1 * s2);
}

// --------------------------- wave-level reductions --------------------------
__device__ __forceinline__ float red_max16(float x) {
  #pragma unroll
  for (int m = 1; m <= 8; m <<= 1) x = fmaxf(x, __shfl_xor(x, m, 32));
  return x;
}
__device__ __forceinline__ float red_sum16(float x) {
  #pragma unroll
  for (int m = 1; m <= 8; m <<= 1) x += __shfl_xor(x, m, 32);
  return x;
}

// ----------------------------- attention ------------------------------------
__global__ __launch_bounds__(256)
void attn_kernel(const __bf16* __restrict__ q_ws, const __bf16* __restrict__ k_ws,
                 const __bf16* __restrict__ v_ws, float* __restrict__ probs,
                 __bf16* __restrict__ attn_out) {
  const int q0 = blockIdx.x * 16;
  const int h  = blockIdx.y;
  const int b  = blockIdx.z;
  const int hk = h / GROUPS_;
  const int wv   = threadIdx.x >> 5;
  const int lane = threadIdx.x & 31;
  const int l15  = lane & 15;
  const int lh   = lane >> 4;

  const int ldq = NH_ * HD_;
  const int ldk = NKV_ * HD_;
  const __bf16* Qb = q_ws + (size_t)b * S_ * NH_ * HD_ + (size_t)h * HD_;
  const __bf16* Kb = k_ws + (size_t)b * S_ * NKV_ * HD_ + (size_t)hk * HD_;
  const __bf16* Vb = v_ws + (size_t)b * S_ * NKV_ * HD_ + (size_t)hk * HD_;

  __shared__ float lds_ms[8][16][2];
  __shared__ float lds_fin[16][2];
  __shared__ float lds_p[8][16 * 32];
  __shared__ float lds_acc[16 * 128];

  v16bf qf[4];
  #pragma unroll
  for (int t = 0; t < 4; ++t) qf[t] = load_a16x32(Qb, ldq, q0, t * 32);

  float m_run[8], s_run[8];
  #pragma unroll
  for (int i = 0; i < 8; ++i) { m_run[i] = -3.0e38f; s_run[i] = 0.0f; }

  for (int blk = wv; blk < S_ / 32; blk += 8) {
    int kb = blk * 32;
    v8f c0 = vzero8(), c1 = vzero8();
    #pragma unroll
    for (int t = 0; t < 4; ++t) {
      v16bf b0 = load_b32x16(Kb, ldk, kb,      t * 32);
      v16bf b1 = load_b32x16(Kb, ldk, kb + 16, t * 32);
      c0 = __builtin_amdgcn_wmma_f32_16x16x32_bf16(false, qf[t], false, b0, (short)0, c0, false, false);
      c1 = __builtin_amdgcn_wmma_f32_16x16x32_bf16(false, qf[t], false, b1, (short)0, c1, false, false);
    }
    #pragma unroll
    for (int i = 0; i < 8; ++i) {
      int row = q0 + lh * 8 + i;
      float v0 = c0[i] * SCALING_ + ((kb + l15      > row) ? NEG_ : 0.0f);
      float v1 = c1[i] * SCALING_ + ((kb + 16 + l15 > row) ? NEG_ : 0.0f);
      float tm = red_max16(fmaxf(v0, v1));
      float mnew = fmaxf(m_run[i], tm);
      float ts = red_sum16(__expf(v0 - mnew) + __expf(v1 - mnew));
      s_run[i] = s_run[i] * __expf(m_run[i] - mnew) + ts;
      m_run[i] = mnew;
    }
  }

  if (l15 == 0) {
    #pragma unroll
    for (int i = 0; i < 8; ++i) {
      int r = lh * 8 + i;
      lds_ms[wv][r][0] = m_run[i];
      lds_ms[wv][r][1] = s_run[i];
    }
  }
  __syncthreads();

  if (threadIdx.x < 16) {
    int r = threadIdx.x;
    float M = -3.0e38f;
    #pragma unroll
    for (int w = 0; w < 8; ++w) M = fmaxf(M, lds_ms[w][r][0]);
    float ssum = 0.0f;
    #pragma unroll
    for (int w = 0; w < 8; ++w) ssum += lds_ms[w][r][1] * __expf(lds_ms[w][r][0] - M);
    lds_fin[r][0] = M;
    lds_fin[r][1] = 1.0f / ssum;
  }
  for (int i = threadIdx.x; i < 16 * 128; i += 256) lds_acc[i] = 0.0f;
  __syncthreads();

  float Mrow[8], Inv[8];
  #pragma unroll
  for (int i = 0; i < 8; ++i) {
    int r = lh * 8 + i;
    Mrow[i] = lds_fin[r][0];
    Inv[i]  = lds_fin[r][1];
  }

  v8f accv[8];
  #pragma unroll
  for (int t = 0; t < 8; ++t) accv[t] = vzero8();

  for (int blk = wv; blk < S_ / 32; blk += 8) {
    int kb = blk * 32;
    v8f c0 = vzero8(), c1 = vzero8();
    #pragma unroll
    for (int t = 0; t < 4; ++t) {
      v16bf b0 = load_b32x16(Kb, ldk, kb,      t * 32);
      v16bf b1 = load_b32x16(Kb, ldk, kb + 16, t * 32);
      c0 = __builtin_amdgcn_wmma_f32_16x16x32_bf16(false, qf[t], false, b0, (short)0, c0, false, false);
      c1 = __builtin_amdgcn_wmma_f32_16x16x32_bf16(false, qf[t], false, b1, (short)0, c1, false, false);
    }
    #pragma unroll
    for (int i = 0; i < 8; ++i) {
      int row = q0 + lh * 8 + i;
      float v0 = c0[i] * SCALING_ + ((kb + l15      > row) ? NEG_ : 0.0f);
      float v1 = c1[i] * SCALING_ + ((kb + 16 + l15 > row) ? NEG_ : 0.0f);
      float p0 = __expf(v0 - Mrow[i]) * Inv[i];
      float p1 = __expf(v1 - Mrow[i]) * Inv[i];
      size_t pb = (((size_t)b * NH_ + h) * S_ + row) * S_;
      probs[pb + kb + l15]      = p0;
      probs[pb + kb + 16 + l15] = p1;
      lds_p[wv][(lh * 8 + i) * 32 + l15]      = p0;
      lds_p[wv][(lh * 8 + i) * 32 + 16 + l15] = p1;
    }
    __builtin_amdgcn_wave_barrier();

    v16bf pa;
    #pragma unroll
    for (int j = 0; j < 8; ++j) {
      pa[j]     = f2bf(lds_p[wv][l15 * 32 + lh * 16 + j]);
      pa[j + 8] = f2bf(lds_p[wv][l15 * 32 + lh * 16 + 8 + j]);
    }
    #pragma unroll
    for (int t = 0; t < 8; ++t) {
      v16bf vf;
      const __bf16* vp = Vb + (size_t)(kb + lh * 16) * ldk + t * 16 + l15;
      #pragma unroll
      for (int j = 0; j < 16; ++j) vf[j] = vp[(size_t)j * ldk];
      accv[t] = __builtin_amdgcn_wmma_f32_16x16x32_bf16(false, pa, false, vf, (short)0, accv[t], false, false);
    }
  }

  for (int wsel = 0; wsel < 8; ++wsel) {
    if (wv == wsel) {
      #pragma unroll
      for (int t = 0; t < 8; ++t)
        #pragma unroll
        for (int i = 0; i < 8; ++i)
          lds_acc[(lh * 8 + i) * 128 + t * 16 + l15] += accv[t][i];
    }
    __syncthreads();
  }

  for (int idx = threadIdx.x; idx < 16 * 128; idx += 256) {
    int r = idx >> 7, d = idx & 127;
    int qq = q0 + r;
    attn_out[(((size_t)b * S_ + qq) * NH_ + h) * HD_ + d] = f2bf(lds_acc[idx]);
  }
}

// ----------------------------------------------------------------------------
extern "C" void kernel_launch(void* const* d_in, const int* in_sizes, int n_in,
                              void* d_out, int out_size, void* d_ws, size_t ws_size,
                              hipStream_t stream) {
  (void)in_sizes; (void)n_in; (void)out_size; (void)ws_size;
  const float* hidden = (const float*)d_in[0];
  const float* cosp   = (const float*)d_in[1];
  const float* sinp   = (const float*)d_in[2];
  // d_in[3] = attention_mask (pure causal; applied analytically in-kernel)
  const float* Wq = (const float*)d_in[4];
  const float* bq = (const float*)d_in[5];
  const float* Wk = (const float*)d_in[6];
  const float* bk = (const float*)d_in[7];
  const float* Wv = (const float*)d_in[8];
  const float* bv = (const float*)d_in[9];
  const float* Wo = (const float*)d_in[10];

  float* out   = (float*)d_out;
  float* probs = out + (size_t)B_ * S_ * H_;

  char* w = (char*)d_ws;
  auto carve = [&](size_t elems) {
    __bf16* p = (__bf16*)w;
    w += elems * sizeof(__bf16);
    return p;
  };
  const size_t nHid = (size_t)B_ * S_ * H_;
  const size_t nWq  = (size_t)NH_ * HD_ * H_;
  const size_t nWkv = (size_t)NKV_ * HD_ * H_;
  const size_t nQ   = (size_t)B_ * S_ * NH_ * HD_;
  const size_t nKV  = (size_t)B_ * S_ * NKV_ * HD_;

  __bf16* hb  = carve(nHid);
  __bf16* Wqb = carve(nWq);
  __bf16* Wkb = carve(nWkv);
  __bf16* Wvb = carve(nWkv);
  __bf16* Wob = carve(nWq);
  __bf16* qb  = carve(nQ);
  __bf16* kb  = carve(nKV);
  __bf16* vb  = carve(nKV);
  __bf16* ab  = carve(nQ);

  // 1) f32 -> bf16 conversions
  cvt_kernel<<<2048, 256, 0, stream>>>(hidden, hb, nHid);
  cvt_kernel<<<2048, 256, 0, stream>>>(Wq, Wqb, nWq);
  cvt_kernel<<<1024, 256, 0, stream>>>(Wk, Wkb, nWkv);
  cvt_kernel<<<1024, 256, 0, stream>>>(Wv, Wvb, nWkv);
  cvt_kernel<<<2048, 256, 0, stream>>>(Wo, Wob, nWq);

  // 2) QKV projections (bf16 WMMA + TDM-staged A, fused bias)
  const int M = B_ * S_;
  gemm_kernel<<<dim3((NH_ * HD_) / 128, M / TILE_M), 256, 0, stream>>>(
      hb, Wqb, bq, qb, nullptr, M, NH_ * HD_, H_);
  gemm_kernel<<<dim3((NKV_ * HD_) / 128, M / TILE_M), 256, 0, stream>>>(
      hb, Wkb, bk, kb, nullptr, M, NKV_ * HD_, H_);
  gemm_kernel<<<dim3((NKV_ * HD_) / 128, M / TILE_M), 256, 0, stream>>>(
      hb, Wvb, bv, vb, nullptr, M, NKV_ * HD_, H_);

  // 3) RoPE on q and k
  {
    long totq = (long)B_ * S_ * NH_ * (HD_ / 2);
    long totk = (long)B_ * S_ * NKV_ * (HD_ / 2);
    rope_kernel<<<(unsigned)((totq + 255) / 256), 256, 0, stream>>>(qb, cosp, sinp, NH_, totq);
    rope_kernel<<<(unsigned)((totk + 255) / 256), 256, 0, stream>>>(kb, cosp, sinp, NKV_, totk);
  }

  // 4) attention (scores + softmax + probs + P*V), flash-style 2-pass
  attn_kernel<<<dim3(S_ / 16, NH_, B_), 256, 0, stream>>>(qb, kb, vb, probs, ab);

  // 5) output projection -> f32 out
  gemm_kernel<<<dim3(H_ / 128, M / TILE_M), 256, 0, stream>>>(
      ab, Wob, nullptr, nullptr, out, M, H_, NH_ * HD_);
}
